// Sub_BidirectionalLSTM_69810398429771
// MI455X (gfx1250) — matrix operand order repaired and checked by reading, not compiled, not run on hardware
//
#include <hip/hip_runtime.h>

// ---------------------------------------------------------------------------
// Problem constants: B=64, T=32, I=512, H=256, O=512, NPRE=25
// ---------------------------------------------------------------------------
#define BB   64
#define TT   32
#define II   512
#define HH   256
#define OO   512
#define G4H  1024      // 4*H
#define NPRE 25

typedef __attribute__((ext_vector_type(16))) __bf16         v16bf;
typedef __attribute__((ext_vector_type(8)))  float          v8f;
typedef __attribute__((ext_vector_type(4)))  unsigned int   v4u;
typedef __attribute__((ext_vector_type(4)))  float          v4f;
typedef __attribute__((ext_vector_type(4)))  unsigned short v4h;

union FragU { v4u q[2]; v16bf v; };

__device__ __forceinline__ unsigned short f2bf(float f) {
  unsigned u = __float_as_uint(f);
  u += 0x7FFFu + ((u >> 16) & 1u);      // round-to-nearest-even
  return (unsigned short)(u >> 16);
}

__device__ __forceinline__ float sigmoidf_(float x) {
  return 1.0f / (1.0f + __expf(-x));
}

__device__ __forceinline__ v8f wmma_bf16(v16bf a, v16bf b, v8f c) {
  // v_wmma_f32_16x16x32_bf16: D = A(16x32) * B(32x16) + C(16x16 f32)
  return __builtin_amdgcn_wmma_f32_16x16x32_bf16(
      /*neg_a=*/false, a, /*neg_b=*/false, b,
      /*c_mod=*/(short)0, c, /*reuse_a=*/false, /*reuse_b=*/false);
}

// A fragment (16x32, bf16): lane<16 holds M=lane, K={kc*32+0..7} and {kc*32+16..23};
// lane>=16 holds M=lane-16, K={kc*32+8..15} and {kc*32+24..31}.
__device__ __forceinline__ v16bf load_a_frag(const unsigned short* base, int ld, int kc) {
  int lane = threadIdx.x & 31;
  int r  = lane & 15;
  int hi = lane >> 4;
  const unsigned short* p = base + (size_t)r * ld + kc * 32 + hi * 8;
  FragU f;
  f.q[0] = *reinterpret_cast<const v4u*>(p);
  f.q[1] = *reinterpret_cast<const v4u*>(p + 16);
  return f.v;
}

// B fragment (32x16, bf16, B = W^T): lane<16 holds N=lane, K={kc*32+0..7} and {kc*32+16..23};
// lane>=16 holds N=lane-16, K={kc*32+8..15} and {kc*32+24..31}. base points at W row block (row-major, ld=K).
__device__ __forceinline__ v16bf load_b_frag(const unsigned short* base, int ld, int kc) {
  int lane = threadIdx.x & 31;
  int n  = lane & 15;
  int hi = lane >> 4;
  const unsigned short* p = base + (size_t)n * ld + kc * 32 + hi * 8;
  FragU f;
  f.q[0] = *reinterpret_cast<const v4u*>(p);
  f.q[1] = *reinterpret_cast<const v4u*>(p + 16);
  return f.v;
}

// ---------------------------------------------------------------------------
// Phase 0: fp32 -> bf16 conversion, 4 elements/thread (b128 in, b64 out)
// ---------------------------------------------------------------------------
__global__ void f32_to_bf16_kernel(const float* __restrict__ in,
                                   unsigned short* __restrict__ out, int n4) {
  int i = blockIdx.x * blockDim.x + threadIdx.x;
  if (i < n4) {
    v4f v = *reinterpret_cast<const v4f*>(in + 4 * (size_t)i);
    v4h o;
    o.x = f2bf(v.x); o.y = f2bf(v.y); o.z = f2bf(v.z); o.w = f2bf(v.w);
    *reinterpret_cast<v4h*>(out + 4 * (size_t)i) = o;
  }
}

// ---------------------------------------------------------------------------
// Generic WMMA GEMM: C[M,N] = A[M,K](bf16, row stride lda) @ W[N,K]^T + bias
// Block = 256 threads (8 waves); block tile 64x128; wave tile 16x64.
// K is a template parameter: fully-unrolled k-loop with positional (kc&1)
// ping-pong buffers -> no register copies, loads hoisted a full k-chunk
// ahead of their consuming WMMAs. 256-thread blocks leave ample VGPRs.
// ---------------------------------------------------------------------------
template <int K>
__global__ __launch_bounds__(256)
void gemm_bf16_kernel(const unsigned short* __restrict__ A, int lda,
                      const unsigned short* __restrict__ W,
                      const float* __restrict__ bias,
                      float* __restrict__ C, int N) {
  const int wave = threadIdx.x >> 5;
  const int lane = threadIdx.x & 31;
  const int mt = wave >> 1;                 // 0..3
  const int nh = wave & 1;                  // 0..1
  const int rowBase = blockIdx.y * 64 + mt * 16;
  const int colBase = blockIdx.x * 128 + nh * 64;

  v8f acc[4];
  v8f zero = {0.f, 0.f, 0.f, 0.f, 0.f, 0.f, 0.f, 0.f};
#pragma unroll
  for (int j = 0; j < 4; ++j) acc[j] = zero;

  const unsigned short* Arow = A + (size_t)rowBase * lda;
  const unsigned short* Wcol[4];
#pragma unroll
  for (int j = 0; j < 4; ++j) Wcol[j] = W + (size_t)(colBase + j * 16) * K;

  constexpr int nkc = K >> 5;

  v16bf a[2];
  v16bf b[2][4];
  a[0] = load_a_frag(Arow, lda, 0);
#pragma unroll
  for (int j = 0; j < 4; ++j) b[0][j] = load_b_frag(Wcol[j], K, 0);

#pragma unroll
  for (int kc = 0; kc < nkc; ++kc) {
    const int cur = kc & 1;
    const int nxt = cur ^ 1;
    if (kc + 1 < nkc) {
      a[nxt] = load_a_frag(Arow, lda, kc + 1);
#pragma unroll
      for (int j = 0; j < 4; ++j) b[nxt][j] = load_b_frag(Wcol[j], K, kc + 1);
    }
#pragma unroll
    for (int j = 0; j < 4; ++j) acc[j] = wmma_bf16(a[cur], b[cur][j], acc[j]);
  }

  // D layout: vgpr r -> M = r + 8*(lane>=16); N = lane&15
  const int r0 = lane >> 4;
  const int n0 = lane & 15;
#pragma unroll
  for (int j = 0; j < 4; ++j) {
    int col = colBase + j * 16 + n0;
    float bv = bias[col];
#pragma unroll
    for (int r = 0; r < 8; ++r) {
      int row = rowBase + r0 * 8 + r;
      C[(size_t)row * N + col] = acc[j][r] + bv;
    }
  }
}

// ---------------------------------------------------------------------------
// Phase 2: 56 independent LSTM chains, one per workgroup.
//   chain 0       : full forward (layer1 f) -> cummax snapshots -> prefix_pool[:, :, 0:H]
//   chain 1       : full backward(layer1 b) -> cummax snapshots -> suffix_pool[:, :, H:2H]
//   chains 2..26  : backward scan over t=len-1..0  -> prefix_pool[len-1][:, H:2H]
//   chains 27..51 : forward  scan over t=len..31   -> suffix_pool[len-1][:, 0:H]
//   chains 52..55 : layer-2 fwd/bwd on x halves    -> r4 bf16 columns (4 slabs of H)
// 1024 threads = 32 waves. Wave w: rows [32*(w>>4), +32), hidden cols [16*(w&15), +16),
// covering all 4 gate columns for those hidden units (register-local cell update).
// Inner GEMM: fully unrolled with a one-deep B ring. In unrolled code the ring
// rotation is SSA-renamed (no copies) and the shallow depth keeps the live
// fragment set inside the per-wave VGPR budget at 32 waves/WG (no spills).
// ---------------------------------------------------------------------------
#define LDS_STRIDE 264   // 256 + 8 halves of pad (528B rows: 16B aligned, bank-skewed)

__global__ __launch_bounds__(1024)
void lstm_chains_kernel(const float* __restrict__ xgf, const float* __restrict__ xgb,
                        const float* __restrict__ xg2f_lo, const float* __restrict__ xg2b_lo,
                        const float* __restrict__ xg2f_hi, const float* __restrict__ xg2b_hi,
                        const unsigned short* __restrict__ whf,
                        const unsigned short* __restrict__ whb,
                        const unsigned short* __restrict__ w2hf,
                        const unsigned short* __restrict__ w2hb,
                        unsigned short* __restrict__ r4_bf,   // [B*T, 1024] bf16
                        float* __restrict__ prefix_pool,      // [25, B, 512]
                        float* __restrict__ suffix_pool) {    // [25, B, 512]
  __shared__ __align__(16) unsigned short h_lds[BB * LDS_STRIDE];

  const int chain = blockIdx.x;
  int dir, t0, nsteps, mode, col_off = 0;
  const float* xg;
  const unsigned short* W;
  float* pool_out = nullptr;

  if (chain == 0) {
    dir = 0; t0 = 0; nsteps = TT; xg = xgf; W = whf; mode = 0;
  } else if (chain == 1) {
    dir = 1; t0 = TT - 1; nsteps = TT; xg = xgb; W = whb; mode = 1;
  } else if (chain <= 26) {
    int len = chain - 1;                       // 1..25
    dir = 1; t0 = len - 1; nsteps = len; xg = xgb; W = whb; mode = 2;
    pool_out = prefix_pool + (size_t)(len - 1) * BB * 2 * HH + HH;   // backward half
  } else if (chain <= 51) {
    int len = chain - 26;                      // 1..25
    dir = 0; t0 = len; nsteps = TT - len; xg = xgf; W = whf; mode = 2;
    pool_out = suffix_pool + (size_t)(len - 1) * BB * 2 * HH;        // forward half
  } else {
    int c2 = chain - 52;                       // 0..3
    dir = c2 & 1; t0 = dir ? (TT - 1) : 0; nsteps = TT; mode = 3;
    col_off = c2 * HH;
    xg = (c2 == 0) ? xg2f_lo : (c2 == 1) ? xg2b_lo : (c2 == 2) ? xg2f_hi : xg2b_hi;
    W = dir ? w2hb : w2hf;
  }

  const int wave = threadIdx.x >> 5;
  const int lane = threadIdx.x & 31;
  const int mg = wave >> 4;        // 0..1: rows 32*mg .. +32
  const int nt = wave & 15;        // 0..15: hidden cols 16*nt .. +16
  const int r0 = lane >> 4;
  const int n0 = lane & 15;
  const int hc = nt * 16 + n0;     // owned hidden column

  // W row blocks for this wave's 4 gate tiles (loop-invariant)
  const unsigned short* Wg[4];
#pragma unroll
  for (int g = 0; g < 4; ++g) Wg[g] = W + (size_t)(g * HH + nt * 16) * HH;
  const unsigned short* hrow0 = h_lds + (size_t)(mg * 32) * LDS_STRIDE;
  const unsigned short* hrow1 = h_lds + (size_t)(mg * 32 + 16) * LDS_STRIDE;

  for (int i = threadIdx.x; i < BB * LDS_STRIDE; i += blockDim.x) h_lds[i] = 0;

  v8f zero = {0.f, 0.f, 0.f, 0.f, 0.f, 0.f, 0.f, 0.f};
  v8f neg  = {-3.4e38f, -3.4e38f, -3.4e38f, -3.4e38f,
              -3.4e38f, -3.4e38f, -3.4e38f, -3.4e38f};
  v8f cfrag[2] = {zero, zero};
  v8f poolf[2] = {neg, neg};

  __syncthreads();

  for (int s = 0; s < nsteps; ++s) {
    const int t = dir ? (t0 - s) : (t0 + s);

    v8f acc[2][4];
#pragma unroll
    for (int mi = 0; mi < 2; ++mi)
#pragma unroll
      for (int g = 0; g < 4; ++g) acc[mi][g] = zero;

    // z = h @ W^T, K = 256 in 8 chunks of 32. One-fragment B ring; fully
    // unrolled so the rotation is SSA-renamed (no copies, no spills).
    {
      v16bf b_cur = load_b_frag(Wg[0], HH, 0);
#pragma unroll
      for (int kc = 0; kc < 8; ++kc) {
        v16bf a0 = load_a_frag(hrow0, LDS_STRIDE, kc);
        v16bf a1 = load_a_frag(hrow1, LDS_STRIDE, kc);
#pragma unroll
        for (int g = 0; g < 4; ++g) {
          v16bf b_nxt;
          if (g < 3)       b_nxt = load_b_frag(Wg[g + 1], HH, kc);
          else if (kc < 7) b_nxt = load_b_frag(Wg[0], HH, kc + 1);
          acc[0][g] = wmma_bf16(a0, b_cur, acc[0][g]);
          acc[1][g] = wmma_bf16(a1, b_cur, acc[1][g]);
          if (g < 3 || kc < 7) b_cur = b_nxt;
        }
      }
    }

    __syncthreads();   // all reads of h_lds for this step done

#pragma unroll
    for (int mi = 0; mi < 2; ++mi) {
#pragma unroll
      for (int r = 0; r < 8; ++r) {
        const int b = mg * 32 + mi * 16 + r0 * 8 + r;
        const float* gp = xg + ((size_t)(b * TT + t)) * G4H + hc;
        float zi = acc[mi][0][r] + gp[0];
        float zf = acc[mi][1][r] + gp[HH];
        float zg = acc[mi][2][r] + gp[2 * HH];
        float zo = acc[mi][3][r] + gp[3 * HH];
        float c2 = sigmoidf_(zf) * cfrag[mi][r] + sigmoidf_(zi) * tanhf(zg);
        float h2 = sigmoidf_(zo) * tanhf(c2);
        cfrag[mi][r] = c2;
        poolf[mi][r] = fmaxf(poolf[mi][r], h2);
        unsigned short hb = f2bf(h2);
        h_lds[(size_t)b * LDS_STRIDE + hc] = hb;
        if (mode == 3)
          r4_bf[((size_t)(b * TT + t)) * G4H + col_off + hc] = hb;
      }
    }

    if (mode == 0 && t < NPRE) {
      float* pp = prefix_pool + (size_t)t * BB * 2 * HH;               // forward half
#pragma unroll
      for (int mi = 0; mi < 2; ++mi)
#pragma unroll
        for (int r = 0; r < 8; ++r) {
          int b = mg * 32 + mi * 16 + r0 * 8 + r;
          pp[(size_t)b * 2 * HH + hc] = poolf[mi][r];
        }
    } else if (mode == 1 && t >= 1 && t <= NPRE) {
      float* sp = suffix_pool + (size_t)(t - 1) * BB * 2 * HH + HH;    // backward half
#pragma unroll
      for (int mi = 0; mi < 2; ++mi)
#pragma unroll
        for (int r = 0; r < 8; ++r) {
          int b = mg * 32 + mi * 16 + r0 * 8 + r;
          sp[(size_t)b * 2 * HH + hc] = poolf[mi][r];
        }
    }

    __syncthreads();   // new h visible before next step's reads
  }

  if (mode == 2) {
#pragma unroll
    for (int mi = 0; mi < 2; ++mi)
#pragma unroll
      for (int r = 0; r < 8; ++r) {
        int b = mg * 32 + mi * 16 + r0 * 8 + r;
        pool_out[(size_t)b * 2 * HH + hc] = poolf[mi][r];
      }
  }
}

// ---------------------------------------------------------------------------
// Host launcher
// ---------------------------------------------------------------------------
extern "C" void kernel_launch(void* const* d_in, const int* in_sizes, int n_in,
                              void* d_out, int out_size, void* d_ws, size_t ws_size,
                              hipStream_t stream) {
  (void)in_sizes; (void)n_in; (void)out_size; (void)ws_size;

  const float* x    = (const float*)d_in[0];
  const float* wif  = (const float*)d_in[1];
  const float* whf  = (const float*)d_in[2];
  const float* bf   = (const float*)d_in[3];
  const float* wib  = (const float*)d_in[4];
  const float* whb  = (const float*)d_in[5];
  const float* bb   = (const float*)d_in[6];
  const float* w2if = (const float*)d_in[7];
  const float* w2hf = (const float*)d_in[8];
  const float* b2f  = (const float*)d_in[9];
  const float* w2ib = (const float*)d_in[10];
  const float* w2hb = (const float*)d_in[11];
  const float* b2b  = (const float*)d_in[12];
  const float* wl   = (const float*)d_in[13];
  const float* bl   = (const float*)d_in[14];

  char* ws = (char*)d_ws;
  size_t off = 0;
  auto alloc = [&](size_t bytes) -> void* {
    void* p = ws + off;
    off += (bytes + 255) & ~(size_t)255;
    return p;
  };

  const int MBT = BB * TT;                                  // 2048 rows
  unsigned short* x_bf    = (unsigned short*)alloc((size_t)MBT * II * 2);
  unsigned short* wif_bf  = (unsigned short*)alloc((size_t)G4H * II * 2);
  unsigned short* wib_bf  = (unsigned short*)alloc((size_t)G4H * II * 2);
  unsigned short* whf_bf  = (unsigned short*)alloc((size_t)G4H * HH * 2);
  unsigned short* whb_bf  = (unsigned short*)alloc((size_t)G4H * HH * 2);
  unsigned short* w2if_bf = (unsigned short*)alloc((size_t)G4H * HH * 2);
  unsigned short* w2ib_bf = (unsigned short*)alloc((size_t)G4H * HH * 2);
  unsigned short* w2hf_bf = (unsigned short*)alloc((size_t)G4H * HH * 2);
  unsigned short* w2hb_bf = (unsigned short*)alloc((size_t)G4H * HH * 2);
  unsigned short* wl_bf   = (unsigned short*)alloc((size_t)OO * G4H * 2);
  unsigned short* r4_bf   = (unsigned short*)alloc((size_t)MBT * G4H * 2);
  float* xgf     = (float*)alloc((size_t)MBT * G4H * 4);
  float* xgb     = (float*)alloc((size_t)MBT * G4H * 4);
  float* xg2f_lo = (float*)alloc((size_t)MBT * G4H * 4);
  float* xg2b_lo = (float*)alloc((size_t)MBT * G4H * 4);
  float* xg2f_hi = (float*)alloc((size_t)MBT * G4H * 4);
  float* xg2b_hi = (float*)alloc((size_t)MBT * G4H * 4);

  auto conv = [&](const float* in, unsigned short* out, int n) {
    int n4 = n >> 2;
    f32_to_bf16_kernel<<<(n4 + 255) / 256, 256, 0, stream>>>(in, out, n4);
  };
  conv(x,    x_bf,    MBT * II);
  conv(wif,  wif_bf,  G4H * II);
  conv(wib,  wib_bf,  G4H * II);
  conv(whf,  whf_bf,  G4H * HH);
  conv(whb,  whb_bf,  G4H * HH);
  conv(w2if, w2if_bf, G4H * HH);
  conv(w2ib, w2ib_bf, G4H * HH);
  conv(w2hf, w2hf_bf, G4H * HH);
  conv(w2hb, w2hb_bf, G4H * HH);
  conv(wl,   wl_bf,   OO * G4H);

  // Phase 1: input-gate GEMMs (bias added in-kernel)
  dim3 blk(256);
  dim3 grd_xg(G4H / 128, MBT / 64);                         // (8, 32)
  gemm_bf16_kernel<II><<<grd_xg, blk, 0, stream>>>(x_bf,      II, wif_bf,  bf,  xgf,     G4H);
  gemm_bf16_kernel<II><<<grd_xg, blk, 0, stream>>>(x_bf,      II, wib_bf,  bb,  xgb,     G4H);
  gemm_bf16_kernel<HH><<<grd_xg, blk, 0, stream>>>(x_bf,      II, w2if_bf, b2f, xg2f_lo, G4H);
  gemm_bf16_kernel<HH><<<grd_xg, blk, 0, stream>>>(x_bf,      II, w2ib_bf, b2b, xg2b_lo, G4H);
  gemm_bf16_kernel<HH><<<grd_xg, blk, 0, stream>>>(x_bf + HH, II, w2if_bf, b2f, xg2f_hi, G4H);
  gemm_bf16_kernel<HH><<<grd_xg, blk, 0, stream>>>(x_bf + HH, II, w2ib_bf, b2b, xg2b_hi, G4H);

  // Phase 2: 56 recurrent chains (pools written straight to d_out)
  float* out         = (float*)d_out;
  float* prefix_pool = out + (size_t)MBT * OO;              // [25, 64, 512]
  float* suffix_pool = prefix_pool + (size_t)NPRE * BB * 2 * HH;
  lstm_chains_kernel<<<56, 1024, 0, stream>>>(
      xgf, xgb, xg2f_lo, xg2b_lo, xg2f_hi, xg2b_hi,
      whf_bf, whb_bf, w2hf_bf, w2hb_bf,
      r4_bf, prefix_pool, suffix_pool);

  // Phase 3: output = r4 @ wl^T + bl
  dim3 grd_out(OO / 128, MBT / 64);                         // (4, 32)
  gemm_bf16_kernel<G4H><<<grd_out, blk, 0, stream>>>(r4_bf, G4H, wl_bf, bl, out, OO);
}
